// CausalSelfAttention_71373766525258
// MI455X (gfx1250) — compile-verified, matrix operand order checked by reading, not run on hardware
//
#include <hip/hip_runtime.h>

// ---------------------------------------------------------------------------
// CDNA5 (gfx1250) causal self-attention forward.
//  x:(2,2048,2048) f32, W_qkv:(6144,2048) f32, W_proj:(2048,2048) f32
//  out:(2,2048,2048) f32
// Pipeline: f32->bf16 convert, WMMA GEMM (QKV -> head-major Q/K/V), RoPE,
// flash-attention (WMMA + online softmax), WMMA GEMM (proj) -> f32 out.
// LDS tiles staged with the Tensor Data Mover (tensor_load_to_lds, TENSORcnt),
// double-buffered: DMA of tile i+1 overlaps WMMA on tile i.
// ---------------------------------------------------------------------------

typedef __bf16 bf16;
typedef unsigned int u32;
typedef unsigned long long u64;
typedef __attribute__((ext_vector_type(16))) bf16  v16bf;
typedef __attribute__((ext_vector_type(8)))  float v8f;
typedef __attribute__((ext_vector_type(4)))  u32   v4u;
typedef __attribute__((ext_vector_type(8)))  int   v8i;
typedef __attribute__((ext_vector_type(4)))  int   v4i;

union FragU { uint4 u[2]; v16bf v; };
union V4H  { uint4 u; bf16 h[8]; };

#define WMMA_BF16(a, b, c) \
  __builtin_amdgcn_wmma_f32_16x16x32_bf16(false, (a), false, (b), (short)0, (c), false, false)

#define NEG_MIN   (-3.4028234663852886e38f)
#define SM_SCALE  (0.08838834764831845f)   // 1/sqrt(128)

// ---------------------------------------------------------------------------
// Tensor Data Mover: 2D tile (elements of 2 bytes) global -> LDS.
// D# packing per CDNA5 ISA:
//   group0: count=1 | lds_addr | global_addr | type=2
//   group1: data_size=1(2B), pad ctrl, tensor_dim0/1, tile_dim0/1, dim0 stride
// All arguments wave-uniform -> SGPR descriptors. Tracked by TENSORcnt.
// Toolchain uses the 6-arg builtin: (v4u, v8i, v4i, v4i, v8i, i32 cpol).
// ---------------------------------------------------------------------------
static __device__ __forceinline__ void tdm_load_2d_bf16(
    const bf16* gaddr, u32 lds_addr,
    u32 tile_d0, u32 tile_d1, u64 stride0_elems,
    u32 pad_en, u32 pad_interval, u32 pad_amount)
{
  u64 ga = (u64)(uintptr_t)gaddr;
  v4u g0;
  g0.x = 0x1u;                                        // count=1, user mode
  g0.y = lds_addr;                                    // LDS byte address
  g0.z = (u32)(ga & 0xffffffffu);                     // global_addr[31:0]
  g0.w = (u32)((ga >> 32) & 0x01ffffffu) | (2u << 30); // [56:32] | type=2

  u32 td0 = tile_d0;          // tensor_dim0 == tile_dim0 (tile fully in-bounds)
  u32 td1 = tile_d1;
  u32 dw0 = (1u << 16)                      // data_size = 1 -> 2 bytes
          | (pad_en << 20)
          | (pad_interval << 22)
          | (pad_amount << 25);
  u32 dw1 = (td0 & 0xffffu) << 16;          // tensor_dim0[15:0] at bit 48
  u32 dw2 = (td0 >> 16) | ((td1 & 0xffffu) << 16);
  u32 dw3 = (td1 >> 16) | (tile_d0 << 16);  // tile_dim0 at bit 112
  u32 dw4 = tile_d1 & 0xffffu;              // tile_dim1; tile_dim2 = 0
  u32 dw5 = (u32)(stride0_elems & 0xffffffffu);
  u32 dw6 = (u32)((stride0_elems >> 32) & 0xffffu); // dim1_stride unused (2D)
  v8i g1 = { (int)dw0, (int)dw1, (int)dw2, (int)dw3,
             (int)dw4, (int)dw5, (int)dw6, 0 };
  v4i z4 = { 0, 0, 0, 0 };
  v8i z8 = { 0, 0, 0, 0, 0, 0, 0, 0 };
  __builtin_amdgcn_tensor_load_to_lds(g0, g1, z4, z4, z8, 0);
}

static __device__ __forceinline__ u32 lds_off(const void* p) {
  return (u32)(uintptr_t)p;   // LDS aperture: low 32 bits are the LDS offset
}

// ---------------------------------------------------------------------------
// fp32 -> bf16 conversion (vectorized, n multiple of 4)
// ---------------------------------------------------------------------------
__global__ __launch_bounds__(256) void cvt_f32_bf16(const float* __restrict__ in,
                                                    bf16* __restrict__ out, int n4) {
  int i = blockIdx.x * 256 + threadIdx.x;
  if (i >= n4) return;
  float4 f = ((const float4*)in)[i];
  union { bf16 h[4]; uint2 u; } o;
  o.h[0] = (bf16)f.x; o.h[1] = (bf16)f.y; o.h[2] = (bf16)f.z; o.h[3] = (bf16)f.w;
  ((uint2*)out)[i] = o.u;
}

// ---------------------------------------------------------------------------
// Tiled WMMA GEMM:  C[m,n] = sum_k A[m,k] * B[n,k]   (A: MxK, B: NxK row-major)
// Block tile 128x128, BK=32, 8 waves -> wave tile 32(M) x 64(N).
// K-tiles staged by TDM, double-buffered, one barrier per iteration.
// mode 0: scatter bf16 into head-major Q/K/V; mode 1: f32 to Cf.
// ---------------------------------------------------------------------------
#define GBK 32
#define GLD 40   // LDS row stride in halves (32 + 8 pad via TDM pad feature)

__global__ __launch_bounds__(256) void gemm_wmma_bf16(
    const bf16* __restrict__ A, const bf16* __restrict__ B,
    float* __restrict__ Cf,
    bf16* __restrict__ Qo, bf16* __restrict__ Ko, bf16* __restrict__ Vo,
    int M, int N, int K, int mode)
{
  __shared__ __align__(16) bf16 As[2][128][GLD];
  __shared__ __align__(16) bf16 Bs[2][128][GLD];

  const int tid  = threadIdx.x;
  const int lane = tid & 31;
  const int wave = tid >> 5;
  const int wm   = wave >> 1;        // 0..3 : 32 rows of M
  const int wn   = wave & 1;         // 0..1 : 64 cols of N
  const int lr   = lane & 15;
  const int lh   = lane >> 4;
  const int k0l  = lh * 8;

  const int bm0 = blockIdx.y * 128;
  const int bn0 = blockIdx.x * 128;

  v8f acc[2][4];
#pragma unroll
  for (int mi = 0; mi < 2; ++mi)
#pragma unroll
    for (int ni = 0; ni < 4; ++ni)
      acc[mi][ni] = (v8f){0.f,0.f,0.f,0.f,0.f,0.f,0.f,0.f};

  const int nk = K / GBK;

  // preload tile 0 (TDM: 128 rows x 32 halves, pad 4 DWORDs every 16 DWORDs)
  if (wave == 0) {
    tdm_load_2d_bf16(&A[(size_t)bm0 * K], lds_off(&As[0][0][0]), GBK, 128, (u64)K, 1, 3, 3);
    tdm_load_2d_bf16(&B[(size_t)bn0 * K], lds_off(&Bs[0][0][0]), GBK, 128, (u64)K, 1, 3, 3);
  }

  for (int i = 0; i < nk; ++i) {
    if (wave == 0) __builtin_amdgcn_s_wait_tensorcnt(0);
    __syncthreads();                         // buffer i&1 ready for all waves

    if (wave == 0 && i + 1 < nk) {           // kick DMA for tile i+1 -> other buf
      int kn = (i + 1) * GBK;
      tdm_load_2d_bf16(&A[(size_t)bm0 * K + kn], lds_off(&As[(i + 1) & 1][0][0]),
                       GBK, 128, (u64)K, 1, 3, 3);
      tdm_load_2d_bf16(&B[(size_t)bn0 * K + kn], lds_off(&Bs[(i + 1) & 1][0][0]),
                       GBK, 128, (u64)K, 1, 3, 3);
    }

    const int buf = i & 1;
    v16bf af[2], bfr[4];
#pragma unroll
    for (int mi = 0; mi < 2; ++mi) {
      const bf16* p = &As[buf][wm * 32 + mi * 16 + lr][0];
      FragU f; f.u[0] = *(const uint4*)(p + k0l); f.u[1] = *(const uint4*)(p + k0l + 16);
      af[mi] = f.v;
    }
#pragma unroll
    for (int ni = 0; ni < 4; ++ni) {
      const bf16* p = &Bs[buf][wn * 64 + ni * 16 + lr][0];
      FragU f; f.u[0] = *(const uint4*)(p + k0l); f.u[1] = *(const uint4*)(p + k0l + 16);
      bfr[ni] = f.v;
    }
#pragma unroll
    for (int mi = 0; mi < 2; ++mi)
#pragma unroll
      for (int ni = 0; ni < 4; ++ni)
        acc[mi][ni] = WMMA_BF16(af[mi], bfr[ni], acc[mi][ni]);
  }

  // epilogue: C layout -> VGPR r holds M = r + 8*lh, N = lr within 16x16 tile
#pragma unroll
  for (int mi = 0; mi < 2; ++mi) {
#pragma unroll
    for (int ni = 0; ni < 4; ++ni) {
#pragma unroll
      for (int r = 0; r < 8; ++r) {
        int row = bm0 + wm * 32 + mi * 16 + r + lh * 8;   // m index (= b*T+t)
        int col = bn0 + wn * 64 + ni * 16 + lr;           // n index (feature f)
        float v = acc[mi][ni][r];
        if (mode == 0) {
          int sel = col >> 11;          // 0=Q 1=K 2=V
          int fr  = col & 2047;
          int h   = fr >> 7;
          int d   = fr & 127;
          int b   = row >> 11;
          int t   = row & 2047;
          size_t off = ((size_t)(b * 16 + h) * 2048 + t) * 128 + d;
          if (sel == 0)      Qo[off] = (bf16)v;
          else if (sel == 1) Ko[off] = (bf16)v;
          else               Vo[off] = (bf16)v;
        } else {
          Cf[(size_t)row * N + col] = v;
        }
      }
    }
  }
}

// ---------------------------------------------------------------------------
// RoPE in-place on head-major bf16 Q and K.
// ---------------------------------------------------------------------------
__global__ __launch_bounds__(256) void rope_kernel(bf16* __restrict__ Q,
                                                   bf16* __restrict__ K, int total) {
  int idx = blockIdx.x * 256 + threadIdx.x;
  if (idx >= total) return;
  int pair = idx & 63;
  int t    = (idx >> 6) & 2047;
  int bh   = idx >> 17;
  size_t base = ((size_t)bh * 2048 + t) * 128 + pair * 2;

  float inv_freq = __expf(-(float)pair * (2.0f / 128.0f) * 9.210340371976184f);
  float ang = (float)t * inv_freq;
  float s, c;
  __sincosf(ang, &s, &c);

  float qe = (float)Q[base], qo = (float)Q[base + 1];
  Q[base]     = (bf16)(qe * c - qo * s);
  Q[base + 1] = (bf16)(qe * s + qo * c);
  float ke = (float)K[base], ko = (float)K[base + 1];
  K[base]     = (bf16)(ke * c - ko * s);
  K[base + 1] = (bf16)(ke * s + ko * c);
}

// ---------------------------------------------------------------------------
// Flash attention (causal). Grid: (T/128, B*H). Block: 256 (8 waves).
// Wave w owns queries [blk*128 + w*16, +16). 32-key K/V tiles in LDS:
// K tile via TDM (double-buffered), V tile manually transposed into the
// *next* buffer while computing (software pipeline, one barrier per tile).
// ---------------------------------------------------------------------------
__global__ __launch_bounds__(256) void flash_attn_wmma(
    const bf16* __restrict__ Q, const bf16* __restrict__ K,
    const bf16* __restrict__ V, bf16* __restrict__ Y)
{
  __shared__ __align__(16) bf16 Ks[2][32][128];     // [key][d]
  __shared__ __align__(16) bf16 Vs[2][128][40];     // transposed [d][key], padded
  __shared__ __align__(16) bf16 Pscr[8][16][32];    // per-wave P transpose scratch

  const int tid  = threadIdx.x;
  const int lane = tid & 31;
  const int wave = tid >> 5;
  const int lr   = lane & 15;
  const int lh   = lane >> 4;
  const int k0l  = lh * 8;

  const int bh = blockIdx.y;            // b*16 + h
  const int b  = bh >> 4;
  const int h  = bh & 15;
  const int qbase = blockIdx.x * 128 + wave * 16;
  const int qlast = qbase + 15;

  const size_t headoff = (size_t)bh * 2048 * 128;
  const bf16* Qh = Q + headoff;
  const bf16* Kh = K + headoff;
  const bf16* Vh = V + headoff;

  // Q fragments: 16 rows x 128 dims = 4 A-frags
  v16bf qf[4];
  {
    const bf16* qrow = Qh + (size_t)(qbase + lr) * 128;
#pragma unroll
    for (int c = 0; c < 4; ++c) {
      FragU f;
      f.u[0] = *(const uint4*)(qrow + c * 32 + k0l);
      f.u[1] = *(const uint4*)(qrow + c * 32 + k0l + 16);
      qf[c] = f.v;
    }
  }

  v8f O[8];
#pragma unroll
  for (int j = 0; j < 8; ++j) O[j] = (v8f){0.f,0.f,0.f,0.f,0.f,0.f,0.f,0.f};
  float mrow[8], lrow[8];
#pragma unroll
  for (int r = 0; r < 8; ++r) { mrow[r] = NEG_MIN; lrow[r] = 0.f; }

  const int nkt = (blockIdx.x + 1) * 4;   // 32-key tiles for this block

  // ---- preload tile 0 ----
  if (wave == 0)   // K tile: 32 rows x 128 halves, contiguous (stride 128)
    tdm_load_2d_bf16(Kh, lds_off(&Ks[0][0][0]), 128, 32, 128, 0, 0, 0);
#pragma unroll
  for (int i2 = tid; i2 < 512; i2 += 256) {   // V tile 0, transposed
    int krow = i2 >> 4, c = (i2 & 15) * 8;
    V4H vv; vv.u = *(const uint4*)&Vh[(size_t)krow * 128 + c];
#pragma unroll
    for (int j = 0; j < 8; ++j) Vs[0][c + j][krow] = vv.h[j];
  }

  for (int it = 0; it < nkt; ++it) {
    if (wave == 0) __builtin_amdgcn_s_wait_tensorcnt(0);
    __syncthreads();                       // tile it (buf it&1) ready

    if (it + 1 < nkt) {                    // pipeline tile it+1 into other buf
      const int nb = (it + 1) & 1;
      const int nk0 = (it + 1) * 32;
      if (wave == 0)
        tdm_load_2d_bf16(Kh + (size_t)nk0 * 128, lds_off(&Ks[nb][0][0]),
                         128, 32, 128, 0, 0, 0);
#pragma unroll
      for (int i2 = tid; i2 < 512; i2 += 256) {
        int krow = i2 >> 4, c = (i2 & 15) * 8;
        V4H vv; vv.u = *(const uint4*)&Vh[(size_t)(nk0 + krow) * 128 + c];
#pragma unroll
        for (int j = 0; j < 8; ++j) Vs[nb][c + j][krow] = vv.h[j];
      }
    }

    const int key0 = it * 32;
    const int buf  = it & 1;
    if (key0 <= qlast) {
      // ---- S = Q * K^T (two 16-key subtiles) ----
      v8f S[2];
#pragma unroll
      for (int s = 0; s < 2; ++s) {
        S[s] = (v8f){0.f,0.f,0.f,0.f,0.f,0.f,0.f,0.f};
#pragma unroll
        for (int c = 0; c < 4; ++c) {
          const bf16* kp = &Ks[buf][s * 16 + lr][0];
          FragU f;
          f.u[0] = *(const uint4*)(kp + c * 32 + k0l);
          f.u[1] = *(const uint4*)(kp + c * 32 + k0l + 16);
          S[s] = WMMA_BF16(qf[c], f.v, S[s]);
        }
      }

      // ---- scale + causal mask ----
      float Sf[2][8];
#pragma unroll
      for (int s = 0; s < 2; ++s) {
        int keyc = key0 + s * 16 + lr;
#pragma unroll
        for (int r = 0; r < 8; ++r) {
          int qr = qbase + r + lh * 8;
          float v = S[s][r] * SM_SCALE;
          Sf[s][r] = (keyc <= qr) ? v : NEG_MIN;
        }
      }

      // ---- online softmax update ----
#pragma unroll
      for (int r = 0; r < 8; ++r) {
        float rm = fmaxf(Sf[0][r], Sf[1][r]);
        rm = fmaxf(rm, __shfl_xor(rm, 1, 32));
        rm = fmaxf(rm, __shfl_xor(rm, 2, 32));
        rm = fmaxf(rm, __shfl_xor(rm, 4, 32));
        rm = fmaxf(rm, __shfl_xor(rm, 8, 32));
        float mn   = fmaxf(mrow[r], rm);
        float corr = __expf(mrow[r] - mn);
        float p0 = __expf(Sf[0][r] - mn);
        float p1 = __expf(Sf[1][r] - mn);
        float rs = p0 + p1;
        rs += __shfl_xor(rs, 1, 32);
        rs += __shfl_xor(rs, 2, 32);
        rs += __shfl_xor(rs, 4, 32);
        rs += __shfl_xor(rs, 8, 32);
        lrow[r] = lrow[r] * corr + rs;
        mrow[r] = mn;
        Sf[0][r] = p0;
        Sf[1][r] = p1;
#pragma unroll
        for (int j = 0; j < 8; ++j) O[j][r] *= corr;
      }

      // ---- transpose P (C layout -> A layout) via per-wave LDS ----
#pragma unroll
      for (int s = 0; s < 2; ++s)
#pragma unroll
        for (int r = 0; r < 8; ++r)
          Pscr[wave][r + lh * 8][s * 16 + lr] = (bf16)Sf[s][r];
      asm volatile("s_wait_dscnt 0x0" ::: "memory");

      v16bf pf;
      {
        const bf16* pp = &Pscr[wave][lr][0];
        FragU f;
        f.u[0] = *(const uint4*)(pp + k0l);
        f.u[1] = *(const uint4*)(pp + k0l + 16);
        pf = f.v;
      }

      // ---- O += P * V ----
#pragma unroll
      for (int j = 0; j < 8; ++j) {
        const bf16* vp = &Vs[buf][j * 16 + lr][0];
        FragU f;
        f.u[0] = *(const uint4*)(vp + k0l);
        f.u[1] = *(const uint4*)(vp + k0l + 16);
        O[j] = WMMA_BF16(pf, f.v, O[j]);
      }
    }
  }

  // ---- finalize ----
  float invl[8];
#pragma unroll
  for (int r = 0; r < 8; ++r) invl[r] = 1.0f / lrow[r];
#pragma unroll
  for (int j = 0; j < 8; ++j) {
#pragma unroll
    for (int r = 0; r < 8; ++r) {
      int trow = qbase + r + lh * 8;
      int col  = h * 128 + j * 16 + lr;
      Y[((size_t)b * 2048 + trow) * 2048 + col] = (bf16)(O[j][r] * invl[r]);
    }
  }
}

// ---------------------------------------------------------------------------
// Launch
// ---------------------------------------------------------------------------
extern "C" void kernel_launch(void* const* d_in, const int* in_sizes, int n_in,
                              void* d_out, int out_size, void* d_ws, size_t ws_size,
                              hipStream_t stream) {
  (void)in_sizes; (void)n_in; (void)out_size; (void)ws_size;

  const float* x     = (const float*)d_in[0];
  const float* Wqkv  = (const float*)d_in[1];
  const float* Wproj = (const float*)d_in[2];
  float*       out   = (float*)d_out;

  const size_t M = 4096;   // B*T
  const size_t E = 2048;
  const size_t F = 6144;   // 3*E

  bf16* xb     = (bf16*)d_ws;          // M*E
  bf16* wqkvb  = xb + M * E;           // F*E
  bf16* wprojb = wqkvb + F * E;        // E*E
  bf16* Qb     = wprojb + E * E;       // M*E (head-major B,H,T,D)
  bf16* Kb     = Qb + M * E;
  bf16* Vb     = Kb + M * E;
  bf16* Yb     = Vb + M * E;           // M*E (B,T,E)

  {
    int n4;
    n4 = (int)(M * E / 4);
    cvt_f32_bf16<<<(n4 + 255) / 256, 256, 0, stream>>>(x, xb, n4);
    n4 = (int)(F * E / 4);
    cvt_f32_bf16<<<(n4 + 255) / 256, 256, 0, stream>>>(Wqkv, wqkvb, n4);
    n4 = (int)(E * E / 4);
    cvt_f32_bf16<<<(n4 + 255) / 256, 256, 0, stream>>>(Wproj, wprojb, n4);
  }

  gemm_wmma_bf16<<<dim3((unsigned)(F / 128), (unsigned)(M / 128)), 256, 0, stream>>>(
      xb, wqkvb, nullptr, Qb, Kb, Vb, (int)M, (int)F, (int)E, 0);

  {
    int total = 2 * 16 * 2048 * 64;
    rope_kernel<<<(total + 255) / 256, 256, 0, stream>>>(Qb, Kb, total);
  }

  flash_attn_wmma<<<dim3(16, 32), 256, 0, stream>>>(Qb, Kb, Vb, Yb);

  gemm_wmma_bf16<<<dim3((unsigned)(E / 128), (unsigned)(M / 128)), 256, 0, stream>>>(
      Yb, wprojb, out, nullptr, nullptr, nullptr, (int)M, (int)E, (int)E, 1);
}
